// KGAT_41686952575548
// MI455X (gfx1250) — compile-verified, matrix operand order checked by reading, not run on hardware
//
#include <hip/hip_runtime.h>
#include <hip/hip_bf16.h>

typedef __attribute__((ext_vector_type(16))) _Float16 v16h;
typedef __attribute__((ext_vector_type(8)))  _Float16 v8h;
typedef __attribute__((ext_vector_type(8)))  float    v8f;

#define NU_    50000
#define NNODE  200000
#define NEDGE  3200000
#define BATCH_ 4096

// ---------------------------------------------------------------------------
// Edge aggregation: hn[target] += feat[neighbor] * value
// One wave32 per edge, 2 dims per lane (float2 gather, 2 hw f32 atomics).
// Feature tables are L2-resident (51 MB < 192 MB L2).
// ---------------------------------------------------------------------------
__global__ __launch_bounds__(256)
void kgat_aggregate(const float* __restrict__ userp, const float* __restrict__ entp,
                    const int* __restrict__ target, const int* __restrict__ neighbor,
                    const float* __restrict__ values, float* __restrict__ hn,
                    int NU, int E)
{
    int e = blockIdx.x * 8 + (threadIdx.x >> 5);
    if (e >= E) return;
    int lane = threadIdx.x & 31;
    int t = target[e];
    int n = neighbor[e];
    float v = values[e];
    const float* src = (n < NU) ? (userp + (size_t)n * 64)
                                : (entp + (size_t)(n - NU) * 64);
    int d = lane * 2;
    float2 f = *(const float2*)(src + d);
    float* dst = hn + (size_t)t * 64 + d;
    __hip_atomic_fetch_add(dst,     f.x * v, __ATOMIC_RELAXED, __HIP_MEMORY_SCOPE_AGENT);
    __hip_atomic_fetch_add(dst + 1, f.y * v, __ATOMIC_RELAXED, __HIP_MEMORY_SCOPE_AGENT);
}

// ---------------------------------------------------------------------------
// Fused dense projection via WMMA:
//   pre = [x + hn | x * hn] (Nx128, f16) @ [W1; W2] (128xNCOLS, f16), f32 acc
//
// Both operands are staged in LDS in *fragment-contiguous* order so every
// v_wmma operand fragment is two ds_load_b128 (no scalar u16 loads, no packs):
//  - A rows swizzled per 32-k block as [0..7,16..23,8..15,24..31]; lane half h
//    reads 16 contiguous halfs at row*128 + ks*32 + h*16 (matches ISA 16-bit
//    A 16x32 layout).
//  - W staged directly in fragment-element order (ks, tile, lane, e).
// pre may alias hn: each block fully stages its rows into LDS before storing.
// ---------------------------------------------------------------------------
template <int NCOLS>
__global__ __launch_bounds__(256)
void kgat_gemm(const float* __restrict__ userp, const float* __restrict__ entp,
               const float* __restrict__ hn, const float* __restrict__ W1,
               const float* __restrict__ W2, float* __restrict__ pre,
               int NU, int NN)
{
    constexpr int NT = NCOLS / 16;
    __shared__ __align__(32) _Float16 Wsw[4 * NT * 512];   // fragment-order weights
    __shared__ __align__(32) _Float16 Alds[128 * 128];     // k-swizzled A rows

    const int tid  = threadIdx.x;
    const int row0 = blockIdx.x * 128;

    // Stage stacked weights [W1; W2] in fragment-element order:
    //   Wsw[(ks*NT + t)*512 + lane*16 + e] = Wcat[k][n]
    //   n = t*16 + (lane&15), h = lane>>4,
    //   k = ks*32 + (e<8 ? h*8+e : 16 + h*8 + (e-8))
    for (int i = tid; i < 4 * NT * 512; i += 256) {
        int ks   = i / (NT * 512);
        int rem  = i - ks * (NT * 512);
        int t    = rem >> 9;
        int r2   = rem & 511;
        int lane = r2 >> 4;
        int e    = r2 & 15;
        int n    = t * 16 + (lane & 15);
        int h    = lane >> 4;
        int k    = ks * 32 + ((e < 8) ? (h * 8 + e) : (16 + h * 8 + (e - 8)));
        float w  = (k < 64) ? W1[k * NCOLS + n] : W2[(k - 64) * NCOLS + n];
        Wsw[i] = (_Float16)w;
    }
    // Stage A = [x + hn | x * hn] rows as f16, k-swizzled within 32-blocks
    for (int i = tid; i < 128 * 64; i += 256) {
        int r = i >> 6, d = i & 63;
        int m = row0 + r;
        float xv = 0.f, hv = 0.f;
        if (m < NN) {
            xv = (m < NU) ? userp[(size_t)m * 64 + d] : entp[(size_t)(m - NU) * 64 + d];
            hv = hn[(size_t)m * 64 + d];
        }
        // element d goes to k = d, element d+64 goes to k = d+64
        int k0 = d, k1 = d + 64;
        int w0 = k0 & 31, g0 = (w0 >> 3) & 3;
        int w1 = k1 & 31, g1 = (w1 >> 3) & 3;
        int p0 = (g0 == 1 || g0 == 2) ? (w0 ^ 24) : w0;
        int p1 = (g1 == 1 || g1 == 2) ? (w1 ^ 24) : w1;
        Alds[r * 128 + (k0 >> 5) * 32 + p0] = (_Float16)(xv + hv);
        Alds[r * 128 + (k1 >> 5) * 32 + p1] = (_Float16)(xv * hv);
    }
    __syncthreads();

    const int wave = tid >> 5, lane = tid & 31;
    const int half = lane >> 4, lrow = lane & 15;
    const int r0   = wave * 16;

    v8f acc[NT] = {};

    const _Float16* arow = &Alds[(r0 + lrow) * 128 + half * 16];
    const _Float16* brow = &Wsw[lane * 16];
#pragma unroll
    for (int ks = 0; ks < 4; ++ks) {
        v8h alo = *(const v8h*)(arow + ks * 32);
        v8h ahi = *(const v8h*)(arow + ks * 32 + 8);
        v16h a  = __builtin_shufflevector(alo, ahi,
                    0,1,2,3,4,5,6,7,8,9,10,11,12,13,14,15);
#pragma unroll
        for (int t = 0; t < NT; ++t) {
            const _Float16* bp = brow + (ks * NT + t) * 512;
            v8h blo = *(const v8h*)(bp);
            v8h bhi = *(const v8h*)(bp + 8);
            v16h b  = __builtin_shufflevector(blo, bhi,
                        0,1,2,3,4,5,6,7,8,9,10,11,12,13,14,15);
            acc[t] = __builtin_amdgcn_wmma_f32_16x16x32_f16(
                false, a, false, b, (short)0, acc[t], false, false);
        }
    }

    // C/D layout: lane holds col lrow of tile t; VGPR r is row r + 8*half
#pragma unroll
    for (int t = 0; t < NT; ++t) {
#pragma unroll
        for (int r = 0; r < 8; ++r) {
            int m = row0 + r0 + half * 8 + r;
            if (m < NN) pre[(size_t)m * NCOLS + t * 16 + lrow] = acc[t][r];
        }
    }
}

// ---------------------------------------------------------------------------
// bias add + leaky_relu(0.01) + row L2-normalize.  One wave32 per node.
// ---------------------------------------------------------------------------
template <int C>
__global__ __launch_bounds__(256)
void kgat_post(const float* __restrict__ pre, const float* __restrict__ b1,
               const float* __restrict__ b2, float* __restrict__ h, int NN)
{
    int node = blockIdx.x * 8 + (threadIdx.x >> 5);
    if (node >= NN) return;
    int lane = threadIdx.x & 31;

    float v0, v1 = 0.f;
    {
        float x = pre[(size_t)node * C + lane] + b1[lane] + b2[lane];
        v0 = (x > 0.f) ? x : 0.01f * x;
    }
    if (C == 64) {
        float x = pre[(size_t)node * C + lane + 32] + b1[lane + 32] + b2[lane + 32];
        v1 = (x > 0.f) ? x : 0.01f * x;
    }
    float s = v0 * v0 + v1 * v1;
#pragma unroll
    for (int o = 16; o > 0; o >>= 1) s += __shfl_xor(s, o, 32);
    float scale = 1.f / fmaxf(sqrtf(s), 1e-12f);
    h[(size_t)node * C + lane] = v0 * scale;
    if (C == 64) h[(size_t)node * C + lane + 32] = v1 * scale;
}

// ---------------------------------------------------------------------------
// Scoring: dot over the 160-wide concat without materializing it.
// One wave32 per batch element; pos and neg together.
// ---------------------------------------------------------------------------
__global__ __launch_bounds__(256)
void kgat_score(const float* __restrict__ user_w, const float* __restrict__ entity_w,
                const float* __restrict__ h1, const float* __restrict__ h2,
                const int* __restrict__ uid, const int* __restrict__ pid,
                const int* __restrict__ nid, float* __restrict__ out, int NU, int B)
{
    int i = blockIdx.x * 8 + (threadIdx.x >> 5);
    if (i >= B) return;
    int lane = threadIdx.x & 31;
    int u = uid[i], p = pid[i], g = nid[i];
    size_t u64 = (size_t)u * 64;
    size_t P = (size_t)(NU + p), G = (size_t)(NU + g);

    float posd = 0.f, negd = 0.f;
    float ux0 = user_w[u64 + lane], ux1 = user_w[u64 + 32 + lane];
    posd += ux0 * entity_w[(size_t)p * 64 + lane] + ux1 * entity_w[(size_t)p * 64 + 32 + lane];
    negd += ux0 * entity_w[(size_t)g * 64 + lane] + ux1 * entity_w[(size_t)g * 64 + 32 + lane];
    float uh0 = h1[u64 + lane], uh1 = h1[u64 + 32 + lane];
    posd += uh0 * h1[P * 64 + lane] + uh1 * h1[P * 64 + 32 + lane];
    negd += uh0 * h1[G * 64 + lane] + uh1 * h1[G * 64 + 32 + lane];
    float u2 = h2[(size_t)u * 32 + lane];
    posd += u2 * h2[P * 32 + lane];
    negd += u2 * h2[G * 32 + lane];

#pragma unroll
    for (int o = 16; o > 0; o >>= 1) {
        posd += __shfl_xor(posd, o, 32);
        negd += __shfl_xor(negd, o, 32);
    }
    if (lane == 0) {
        out[i]     = posd;
        out[B + i] = negd;
    }
}

// ---------------------------------------------------------------------------
extern "C" void kernel_launch(void* const* d_in, const int* in_sizes, int n_in,
                              void* d_out, int out_size, void* d_ws, size_t ws_size,
                              hipStream_t stream)
{
    const float* user_w   = (const float*)d_in[0];
    const float* entity_w = (const float*)d_in[1];
    const float* W1a = (const float*)d_in[2];
    const float* b1a = (const float*)d_in[3];
    const float* W2a = (const float*)d_in[4];
    const float* b2a = (const float*)d_in[5];
    const float* W1b = (const float*)d_in[6];
    const float* b1b = (const float*)d_in[7];
    const float* W2b = (const float*)d_in[8];
    const float* b2b = (const float*)d_in[9];
    const float* values   = (const float*)d_in[10];
    const int*   target   = (const int*)d_in[11];
    const int*   neighbor = (const int*)d_in[12];
    const int*   uids = (const int*)d_in[13];
    const int*   pids = (const int*)d_in[14];
    const int*   nids = (const int*)d_in[15];
    float* out = (float*)d_out;

    const int NU = NU_, NN = NNODE, E = NEDGE, B = BATCH_;

    // Workspace layout (~154 MB total):
    //   hn   : NN*64 f32 (aggregation accumulator; pre1 written in place)
    //   h1   : NN*64 f32
    //   pre2 : NN*32 f32
    //   h2   : NN*32 f32
    char*  ws   = (char*)d_ws;
    float* hn   = (float*)ws;
    float* h1   = (float*)(ws + (size_t)NN * 64 * 4);
    float* pre2 = (float*)(ws + (size_t)NN * 64 * 4 * 2);
    float* h2   = pre2 + (size_t)NN * 32;

    const int aggBlocks  = (E + 7) / 8;
    const int gemmBlocks = (NN + 127) / 128;
    const int postBlocks = (NN + 7) / 8;

    // ---- Layer 1 ----
    hipMemsetAsync(hn, 0, (size_t)NN * 64 * 4, stream);
    kgat_aggregate<<<aggBlocks, 256, 0, stream>>>(user_w, entity_w, target, neighbor,
                                                  values, hn, NU, E);
    kgat_gemm<64><<<gemmBlocks, 256, 0, stream>>>(user_w, entity_w, hn, W1a, W2a,
                                                  hn /* in-place pre */, NU, NN);
    kgat_post<64><<<postBlocks, 256, 0, stream>>>(hn, b1a, b2a, h1, NN);

    // ---- Layer 2 (source features = h1 via pointer-split trick) ----
    hipMemsetAsync(hn, 0, (size_t)NN * 64 * 4, stream);
    kgat_aggregate<<<aggBlocks, 256, 0, stream>>>(h1, h1 + (size_t)NU * 64, target, neighbor,
                                                  values, hn, NU, E);
    kgat_gemm<32><<<gemmBlocks, 256, 0, stream>>>(h1, h1 + (size_t)NU * 64, hn, W1b, W2b,
                                                  pre2, NU, NN);
    kgat_post<32><<<postBlocks, 256, 0, stream>>>(pre2, b1b, b2b, h2, NN);

    // ---- Scoring ----
    kgat_score<<<(B + 7) / 8, 256, 0, stream>>>(user_w, entity_w, h1, h2,
                                                uids, pids, nids, out, NU, B);
}
